// Cross_Attention_48584670052665
// MI455X (gfx1250) — matrix operand order repaired and checked by reading, not compiled, not run on hardware
//
#include <hip/hip_runtime.h>
#include <hip/hip_bf16.h>

typedef _Float16 half_t;
typedef __attribute__((ext_vector_type(16))) _Float16 v16h;
typedef __attribute__((ext_vector_type(8)))  _Float16 v8h;
typedef __attribute__((ext_vector_type(8)))  float    v8f;
typedef __attribute__((ext_vector_type(4)))  float    v4f;
typedef __attribute__((ext_vector_type(4)))  unsigned int u32x4;
typedef __attribute__((ext_vector_type(8)))  int      i32x8;
typedef __attribute__((ext_vector_type(4)))  int      i32x4;

#define WMMA_F16(a, b, c) \
    __builtin_amdgcn_wmma_f32_16x16x32_f16(false, (a), false, (b), (short)0, (c), false, false)

static constexpr int DIM   = 512;
static constexpr int NSEQ  = 2048;
static constexpr int BATCH = 2;
static constexpr int HEADS = 8;
static constexpr int DHEAD = 64;
static constexpr size_t QKV_ELEMS = (size_t)BATCH * HEADS * NSEQ * DHEAD;  // 2M half each
static constexpr size_t WPACK_ELEMS = (size_t)DIM * DIM;                    // 256K half per W

// ---------------------------------------------------------------------------
// TDM: issue a 2D tensor_load_to_lds (global -> LDS DMA), 6-arg builtin form
// (this toolchain's signature). D# bitfields per CDNA5 ISA
// §8.3/8.4; groups 2/3 (and the trailing i32x8) zero for a 2D tensor.
// ---------------------------------------------------------------------------
__device__ __forceinline__ void tdm_load_2d(const void* gaddr, unsigned lds_off,
                                            unsigned tensor_d0, unsigned tensor_d1,
                                            unsigned tile_d0, unsigned tile_d1,
                                            unsigned stride0)
{
    const unsigned long long ga = (unsigned long long)gaddr;
    u32x4 g0;
    g0[0] = 1u;                                            // count=1, user descriptor
    g0[1] = lds_off;                                       // lds_addr (bytes)
    g0[2] = (unsigned)(ga & 0xFFFFFFFFu);                  // global_addr[31:0]
    g0[3] = (unsigned)((ga >> 32) & 0x1FFFFFFu)            // global_addr[56:32]
          | (2u << 30);                                    // type = 2 ("image")
    i32x8 g1;
    g1[0] = 0x10000;                                       // data_size=1 -> 2 bytes
    g1[1] = (int)((tensor_d0 & 0xFFFFu) << 16);            // tensor_dim0[15:0]
    g1[2] = (int)((tensor_d0 >> 16) | ((tensor_d1 & 0xFFFFu) << 16));
    g1[3] = (int)((tensor_d1 >> 16) | (tile_d0 << 16));    // tile_dim0
    g1[4] = (int)tile_d1;                                  // tile_dim1 (tile_dim2=0)
    g1[5] = (int)stride0;                                  // tensor_dim0_stride[31:0]
    g1[6] = 0;
    g1[7] = 0;
    const i32x4 z4 = {};
    const i32x8 z8 = {};
    __builtin_amdgcn_tensor_load_to_lds(g0, g1, z4, z4, z8, 0);
}

// ---------------------------------------------------------------------------
// Kernel 0: pack Wq/Wk/Wv/Wo (fp32 [k][n]) into fragment-major f16 so every
// WMMA B-fragment is ONE contiguous 32-byte v16h load.
// ---------------------------------------------------------------------------
__global__ __launch_bounds__(256) void pack_w_kernel(
    const float* __restrict__ Wq, const float* __restrict__ Wk,
    const float* __restrict__ Wv, const float* __restrict__ Wo,
    half_t* __restrict__ Wp)
{
    const int z = blockIdx.y;                                   // 0..3
    const float* W = (z == 0) ? Wq : (z == 1) ? Wk : (z == 2) ? Wv : Wo;
    const int tid   = blockIdx.x * 256 + threadIdx.x;           // 16384 per weight
    const int ntile = tid >> 9;
    const int kstep = (tid >> 5) & 15;
    const int lane  = tid & 31;
    const int lr = lane & 15, lh = lane >> 4;

    v16h frag;
    #pragma unroll
    for (int i = 0; i < 16; ++i)
        frag[i] = (half_t)W[(size_t)(kstep * 32 + lh * 8 + (i < 8 ? i : i + 8)) * DIM + ntile * 16 + lr];
    *(v16h*)(Wp + (size_t)z * WPACK_ELEMS + (size_t)tid * 16) = frag;
}

// ---------------------------------------------------------------------------
// Kernel 1: P = X @ Wpacked. One wave -> 16x64 tile (A reused over 4 WMMAs).
// Q,K stored [b,h,n,d]; V stored TRANSPOSED [b,h,d,n].
// ---------------------------------------------------------------------------
__global__ __launch_bounds__(128) void proj_kernel(
    const float* __restrict__ x, const float* __restrict__ m, const float* __restrict__ y,
    const half_t* __restrict__ Wp, half_t* __restrict__ qkv)
{
    const int z = blockIdx.z;
    const float* X = (z == 0) ? x : (z == 1) ? m : y;
    const half_t* W = Wp + (size_t)z * WPACK_ELEMS;
    half_t* D = qkv + (size_t)z * QKV_ELEMS;
    const float scale = (z == 0) ? 0.04419417382415922f /* 512^-0.5 */ : 1.0f;

    const int lane = threadIdx.x & 31;
    const int wave = threadIdx.x >> 5;
    const int lr = lane & 15, lh = lane >> 4;
    const int t    = blockIdx.x * 4 + wave;
    const int row0 = (t >> 3) * 16;
    const int ng   = t & 7;

    v8f acc0 = {}, acc1 = {}, acc2 = {}, acc3 = {};
    const float* Xrow = X + (size_t)(row0 + lr) * DIM;

    for (int kk = 0; kk < DIM; kk += 32) {
        const v4f f0 = *(const v4f*)(Xrow + kk + lh * 8);
        const v4f f1 = *(const v4f*)(Xrow + kk + lh * 8 + 4);
        const v4f f2 = *(const v4f*)(Xrow + kk + lh * 8 + 16);
        const v4f f3 = *(const v4f*)(Xrow + kk + lh * 8 + 20);
        v16h a;
        #pragma unroll
        for (int i = 0; i < 4; ++i) {
            a[i]      = (half_t)f0[i];
            a[i + 4]  = (half_t)f1[i];
            a[i + 8]  = (half_t)f2[i];
            a[i + 12] = (half_t)f3[i];
        }
        const size_t kb = (size_t)(kk >> 5) * 32 + lane;
        const v16h b0 = *(const v16h*)(W + (((size_t)(ng * 4 + 0) * 16) * 32 + kb) * 16);
        const v16h b1 = *(const v16h*)(W + (((size_t)(ng * 4 + 1) * 16) * 32 + kb) * 16);
        const v16h b2 = *(const v16h*)(W + (((size_t)(ng * 4 + 2) * 16) * 32 + kb) * 16);
        const v16h b3 = *(const v16h*)(W + (((size_t)(ng * 4 + 3) * 16) * 32 + kb) * 16);
        acc0 = WMMA_F16(a, b0, acc0);
        acc1 = WMMA_F16(a, b1, acc1);
        acc2 = WMMA_F16(a, b2, acc2);
        acc3 = WMMA_F16(a, b3, acc3);
    }

    #pragma unroll
    for (int nsub = 0; nsub < 4; ++nsub) {
        const v8f acc = (nsub == 0) ? acc0 : (nsub == 1) ? acc1 : (nsub == 2) ? acc2 : acc3;
        #pragma unroll
        for (int v = 0; v < 8; ++v) {
            const int r = row0 + lh * 8 + v;
            const int c = ng * 64 + nsub * 16 + lr;
            const int bi = r >> 11, n = r & (NSEQ - 1);
            const int h = c >> 6, d = c & (DHEAD - 1);
            const half_t val = (half_t)(acc[v] * scale);
            if (z != 2)
                D[((size_t)(bi * HEADS + h) * NSEQ + n) * DHEAD + d] = val;
            else
                D[((size_t)(bi * HEADS + h) * DHEAD + d) * NSEQ + n] = val;
        }
    }
}

// ---------------------------------------------------------------------------
// Kernel 2: flash attention. All 4 waves of a block share (b,h); K / V^T key
// tiles are DMA'd into LDS by the Tensor Data Mover (wave 0), double-buffered
// and synchronized with s_wait_tensorcnt + workgroup barriers. P staged
// through LDS for the S->A transpose.
// ---------------------------------------------------------------------------
struct AttnSmem {
    half_t k[2][32 * 64];    // [buf][key][d]      4 KB each
    half_t vt[2][64 * 32];   // [buf][d][key]      4 KB each
    half_t p[4][16][32];     // per-wave P staging 4 KB
};

__global__ __launch_bounds__(128) void attn_kernel(
    const half_t* __restrict__ qkv, half_t* __restrict__ attn_out)
{
    const int bh   = blockIdx.y;               // 0..15
    const int wave = threadIdx.x >> 5;
    const int lane = threadIdx.x & 31;
    const int lr = lane & 15, lh = lane >> 4;
    const int q0 = (blockIdx.x * 4 + wave) * 16;

    const half_t* Q  = qkv + (size_t)bh * NSEQ * DHEAD;
    const half_t* K  = qkv + QKV_ELEMS + (size_t)bh * NSEQ * DHEAD;
    const half_t* Vt = qkv + 2 * QKV_ELEMS + (size_t)bh * DHEAD * NSEQ;   // [d][n]

    __shared__ AttnSmem smem;
    const unsigned kOff0  = (unsigned)(size_t)&smem.k[0][0];
    const unsigned kOff1  = (unsigned)(size_t)&smem.k[1][0];
    const unsigned vtOff0 = (unsigned)(size_t)&smem.vt[0][0];
    const unsigned vtOff1 = (unsigned)(size_t)&smem.vt[1][0];

    // Q fragments (A 16x32 layout): contiguous v8h loads, resident all loop
    v16h aq0, aq1;
    {
        const half_t* qrow = Q + (size_t)(q0 + lr) * DHEAD;
        const v8h q00 = *(const v8h*)(qrow + lh * 16);
        const v8h q01 = *(const v8h*)(qrow + lh * 16 + 8);
        const v8h q10 = *(const v8h*)(qrow + 32 + lh * 16);
        const v8h q11 = *(const v8h*)(qrow + 32 + lh * 16 + 8);
        #pragma unroll
        for (int i = 0; i < 8; ++i) {
            aq0[i] = q00[i]; aq0[i + 8] = q01[i];
            aq1[i] = q10[i]; aq1[i + 8] = q11[i];
        }
    }

    // Prologue: prime the 2-deep TDM pipeline (wave 0 issues for the block)
    if (wave == 0) {
        tdm_load_2d(K,               kOff0,  DHEAD, NSEQ, DHEAD, 32, DHEAD);  // keys 0..31
        tdm_load_2d(Vt,              vtOff0, NSEQ, DHEAD, 32, DHEAD, NSEQ);
        tdm_load_2d(K + 32 * DHEAD,  kOff1,  DHEAD, NSEQ, DHEAD, 32, DHEAD);  // keys 32..63
        tdm_load_2d(Vt + 32,         vtOff1, NSEQ, DHEAD, 32, DHEAD, NSEQ);
    }

    v8f o0 = {}, o1 = {}, o2 = {}, o3 = {};
    float rmax[8], rsum[8];
    #pragma unroll
    for (int v = 0; v < 8; ++v) { rmax[v] = -3.0e38f; rsum[v] = 0.0f; }

    for (int jb = 0; jb < NSEQ / 32; ++jb) {
        const int buf = jb & 1;
        if (wave == 0) {     // current pair done; next pair may remain in flight
            if (jb < NSEQ / 32 - 1) __builtin_amdgcn_s_wait_tensorcnt(2);
            else                    __builtin_amdgcn_s_wait_tensorcnt(0);
        }
        __syncthreads();     // LDS tiles valid for all waves

        const half_t* kb = smem.k[buf];
        const half_t* vb = smem.vt[buf];

        // S = Q @ K^T : contiguous LDS fragments
        const v16h bk00 = *(const v16h*)(kb + (size_t)(0  + lr) * DHEAD + lh * 16);
        const v16h bk01 = *(const v16h*)(kb + (size_t)(0  + lr) * DHEAD + 32 + lh * 16);
        const v16h bk10 = *(const v16h*)(kb + (size_t)(16 + lr) * DHEAD + lh * 16);
        const v16h bk11 = *(const v16h*)(kb + (size_t)(16 + lr) * DHEAD + 32 + lh * 16);

        v8f s0 = {}, s1 = {};
        s0 = WMMA_F16(aq0, bk00, s0);  s0 = WMMA_F16(aq1, bk01, s0);
        s1 = WMMA_F16(aq0, bk10, s1);  s1 = WMMA_F16(aq1, bk11, s1);

        // Online softmax (row r = lh*8+v lives across 16 lanes of a half-wave)
        #pragma unroll
        for (int v = 0; v < 8; ++v) {
            float mv = fmaxf(s0[v], s1[v]);
            #pragma unroll
            for (int off = 1; off < 16; off <<= 1)
                mv = fmaxf(mv, __shfl_xor(mv, off, 32));
            const float nm   = fmaxf(rmax[v], mv);
            const float corr = __expf(rmax[v] - nm);
            const float p0 = __expf(s0[v] - nm);
            const float p1 = __expf(s1[v] - nm);
            float rs = p0 + p1;
            #pragma unroll
            for (int off = 1; off < 16; off <<= 1)
                rs += __shfl_xor(rs, off, 32);
            rsum[v] = rsum[v] * corr + rs;
            rmax[v] = nm;
            o0[v] *= corr; o1[v] *= corr; o2[v] *= corr; o3[v] *= corr;
            smem.p[wave][lh * 8 + v][lr]      = (half_t)p0;
            smem.p[wave][lh * 8 + v][lr + 16] = (half_t)p1;
        }

        asm volatile("s_wait_dscnt 0" ::: "memory");   // LDS write -> read, same wave

        v16h ap;
        #pragma unroll
        for (int i = 0; i < 16; ++i)
            ap[i] = smem.p[wave][lr][lh * 16 + i];

        // O += P @ V via V^T tile in LDS
        const v16h bv0 = *(const v16h*)(vb + (size_t)(0  + lr) * 32 + lh * 16);
        const v16h bv1 = *(const v16h*)(vb + (size_t)(16 + lr) * 32 + lh * 16);
        const v16h bv2 = *(const v16h*)(vb + (size_t)(32 + lr) * 32 + lh * 16);
        const v16h bv3 = *(const v16h*)(vb + (size_t)(48 + lr) * 32 + lh * 16);
        o0 = WMMA_F16(ap, bv0, o0);
        o1 = WMMA_F16(ap, bv1, o1);
        o2 = WMMA_F16(ap, bv2, o2);
        o3 = WMMA_F16(ap, bv3, o3);

        __syncthreads();     // everyone done reading buf before DMA overwrites it
        if (wave == 0 && jb + 2 < NSEQ / 32) {
            const int j2 = (jb + 2) * 32;
            tdm_load_2d(K + (size_t)j2 * DHEAD, buf ? kOff1 : kOff0,
                        DHEAD, NSEQ, DHEAD, 32, DHEAD);
            tdm_load_2d(Vt + j2, buf ? vtOff1 : vtOff0,
                        NSEQ, DHEAD, 32, DHEAD, NSEQ);
        }
    }

    // Normalize, store [b, n, h*64+d] f16 for the output projection
    const int bi = bh >> 3, h = bh & 7;
    #pragma unroll
    for (int v = 0; v < 8; ++v) {
        const float inv = 1.0f / rsum[v];
        const int qrow = q0 + lh * 8 + v;
        const size_t base = ((size_t)bi * NSEQ + qrow) * (HEADS * DHEAD) + h * DHEAD + lr;
        attn_out[base +  0] = (half_t)(o0[v] * inv);
        attn_out[base + 16] = (half_t)(o1[v] * inv);
        attn_out[base + 32] = (half_t)(o2[v] * inv);
        attn_out[base + 48] = (half_t)(o3[v] * inv);
    }
}

// ---------------------------------------------------------------------------
// Kernel 3: out = attn(f16) @ Wo_packed + bo  (f32 out); 16x64 tile per wave
// ---------------------------------------------------------------------------
__global__ __launch_bounds__(128) void out_kernel(
    const half_t* __restrict__ A, const half_t* __restrict__ WoP,
    const float* __restrict__ bo, float* __restrict__ out)
{
    const int lane = threadIdx.x & 31;
    const int wave = threadIdx.x >> 5;
    const int lr = lane & 15, lh = lane >> 4;
    const int t    = blockIdx.x * 4 + wave;
    const int row0 = (t >> 3) * 16;
    const int ng   = t & 7;

    v8f acc0 = {}, acc1 = {}, acc2 = {}, acc3 = {};
    const half_t* Arow = A + (size_t)(row0 + lr) * DIM;

    for (int kk = 0; kk < DIM; kk += 32) {
        const v8h a0 = *(const v8h*)(Arow + kk + lh * 8);
        const v8h a1 = *(const v8h*)(Arow + kk + lh * 8 + 16);
        v16h a;
        #pragma unroll
        for (int i = 0; i < 8; ++i) { a[i] = a0[i]; a[i + 8] = a1[i]; }

        const size_t kb = (size_t)(kk >> 5) * 32 + lane;
        const v16h b0 = *(const v16h*)(WoP + (((size_t)(ng * 4 + 0) * 16) * 32 + kb) * 16);
        const v16h b1 = *(const v16h*)(WoP + (((size_t)(ng * 4 + 1) * 16) * 32 + kb) * 16);
        const v16h b2 = *(const v16h*)(WoP + (((size_t)(ng * 4 + 2) * 16) * 32 + kb) * 16);
        const v16h b3 = *(const v16h*)(WoP + (((size_t)(ng * 4 + 3) * 16) * 32 + kb) * 16);
        acc0 = WMMA_F16(a, b0, acc0);
        acc1 = WMMA_F16(a, b1, acc1);
        acc2 = WMMA_F16(a, b2, acc2);
        acc3 = WMMA_F16(a, b3, acc3);
    }

    #pragma unroll
    for (int nsub = 0; nsub < 4; ++nsub) {
        const v8f acc = (nsub == 0) ? acc0 : (nsub == 1) ? acc1 : (nsub == 2) ? acc2 : acc3;
        const int col = ng * 64 + nsub * 16 + lr;
        const float bias = bo[col];
        #pragma unroll
        for (int v = 0; v < 8; ++v)
            out[(size_t)(row0 + lh * 8 + v) * DIM + col] = acc[v] + bias;
    }
}

// ---------------------------------------------------------------------------
extern "C" void kernel_launch(void* const* d_in, const int* in_sizes, int n_in,
                              void* d_out, int out_size, void* d_ws, size_t ws_size,
                              hipStream_t stream) {
    const float* x  = (const float*)d_in[0];
    const float* m  = (const float*)d_in[1];
    const float* y  = (const float*)d_in[2];
    const float* Wq = (const float*)d_in[3];
    const float* Wk = (const float*)d_in[4];
    const float* Wv = (const float*)d_in[5];
    const float* Wo = (const float*)d_in[6];
    const float* bo = (const float*)d_in[7];
    float* out = (float*)d_out;

    const size_t need = (4 * QKV_ELEMS + 4 * WPACK_ELEMS) * sizeof(half_t);  // 18 MB
    if (ws_size < need) return;
    half_t* qkv      = (half_t*)d_ws;              // Q | K | V^T
    half_t* attn_out = qkv + 3 * QKV_ELEMS;
    half_t* Wp       = qkv + 4 * QKV_ELEMS;        // packed Wq|Wk|Wv|Wo

    pack_w_kernel<<<dim3(64, 4), 256, 0, stream>>>(Wq, Wk, Wv, Wo, Wp);
    proj_kernel<<<dim3(512, 1, 3), 128, 0, stream>>>(x, m, y, Wp, qkv);
    attn_kernel<<<dim3(32, 16), 128, 0, stream>>>(qkv, attn_out);
    out_kernel<<<dim3(512), 128, 0, stream>>>(attn_out, Wp + 3 * WPACK_ELEMS, bo, out);
}
